// Deduce_70128226009499
// MI455X (gfx1250) — matrix operand order repaired and checked by reading, not compiled, not run on hardware
//
#include <hip/hip_runtime.h>

// ---------------------------------------------------------------------------
// CDNA5 (gfx1250) adaptive-softmax NLL, fused GEMM + online logsumexp.
// BF16 WMMA (v_wmma_f32_16x16x32_bf16), wave32.
// Register blocking: each wave computes a 32x32 output block (4 WMMA tiles,
// independent accumulators) -> 1 wmma per 2 b128 loads per lane.
// Workspace requirement: ~76 MB.
// ---------------------------------------------------------------------------

typedef __attribute__((ext_vector_type(16))) __bf16       v16bf;
typedef __attribute__((ext_vector_type(8)))  float        v8f;
typedef __attribute__((ext_vector_type(4)))  unsigned int u32x4;

union Frag { v16bf v; u32x4 q[2]; };

#define NTOK   4096
#define HDIM   1024
#define NEG_BIG (-1e30f)

__device__ __forceinline__ unsigned short f2bf(float f) {
  unsigned int u = __float_as_uint(f);
  u += 0x7FFFu + ((u >> 16) & 1u);          // round-to-nearest-even
  return (unsigned short)(u >> 16);
}

// ---------------- tiny utility kernels ----------------
__global__ void k_zero16(unsigned short* p, long long n) {
  for (long long i = (long long)blockIdx.x * 256 + threadIdx.x; i < n;
       i += (long long)gridDim.x * 256) p[i] = 0;
}
__global__ void k_cvt(const float* __restrict__ src, unsigned short* __restrict__ dst,
                      int srcRows, int srcCols, int dstCols, int rowOff) {
  long long n = (long long)srcRows * srcCols;
  for (long long i = (long long)blockIdx.x * 256 + threadIdx.x; i < n;
       i += (long long)gridDim.x * 256) {
    int r = (int)(i / srcCols), c = (int)(i % srcCols);
    dst[(long long)(rowOff + r) * dstCols + c] = f2bf(src[i]);
  }
}
__global__ void k_fill_f32(float* p, float v, int n) {
  int i = blockIdx.x * 256 + threadIdx.x;
  if (i < n) p[i] = v;
}
__global__ void k_copy_f32(float* __restrict__ dst, const float* __restrict__ src, int n) {
  int i = blockIdx.x * 256 + threadIdx.x;
  if (i < n) dst[i] = src[i];
}

// ---------------- adapter GEMM: xt = x @ Aw^T, bf16 out ----------------
// A: [NTOK][K] bf16 row-major.  Bw: [Pp][K] bf16 row-major (rows = out cols).
// Block: 128 threads, 32-token tile; wave computes 32x32 output block.
__global__ __launch_bounds__(128) void gemm_xt(
    const unsigned short* __restrict__ A, int K,
    const unsigned short* __restrict__ Bw, int Pp,
    unsigned short* __restrict__ Out) {
  __shared__ __align__(16) unsigned short sA[32 * HDIM];   // 64 KB

  const int tid  = threadIdx.x;
  const int wave = tid >> 5, lane = tid & 31;
  const int hi = lane >> 4, lc = lane & 15, koff = hi * 8;
  const int tok0 = blockIdx.x * 32;

  { // cooperative 128b copy of the 32 x K bf16 x-tile into LDS
    const u32x4* g = (const u32x4*)(A + (size_t)tok0 * K);
    u32x4* s = (u32x4*)sA;
    const int n = 4 * K;
    for (int i = tid; i < n; i += 128) s[i] = g[i];
  }
  __syncthreads();

  const int col0 = blockIdx.y * 128 + wave * 32 + lc;
  const int col1 = col0 + 16;
  const unsigned short* brow0 = Bw + (size_t)col0 * K;
  const unsigned short* brow1 = Bw + (size_t)col1 * K;
  v8f z = {0.f,0.f,0.f,0.f,0.f,0.f,0.f,0.f};
  v8f acc00 = z, acc01 = z, acc10 = z, acc11 = z;
#pragma unroll 2
  for (int k0 = 0; k0 < K; k0 += 32) {
    Frag a0, a1, b0, b1;
    // A fragments (rows lc and lc+16): K interleave {koff..+7, 16+koff..+7}
    a0.q[0] = *(const u32x4*)(sA + lc * K + k0 + koff);
    a0.q[1] = *(const u32x4*)(sA + lc * K + k0 + 16 + koff);
    a1.q[0] = *(const u32x4*)(sA + (lc + 16) * K + k0 + koff);
    a1.q[1] = *(const u32x4*)(sA + (lc + 16) * K + k0 + 16 + koff);
    // B fragments (cols col0, col1): K interleave {koff..+7, 16+koff..+7}
    b0.q[0] = *(const u32x4*)(brow0 + k0 + koff);
    b0.q[1] = *(const u32x4*)(brow0 + k0 + 16 + koff);
    b1.q[0] = *(const u32x4*)(brow1 + k0 + koff);
    b1.q[1] = *(const u32x4*)(brow1 + k0 + 16 + koff);
    acc00 = __builtin_amdgcn_wmma_f32_16x16x32_bf16(false, a0.v, false, b0.v, (short)0, acc00, false, false);
    acc01 = __builtin_amdgcn_wmma_f32_16x16x32_bf16(false, a0.v, false, b1.v, (short)0, acc01, false, false);
    acc10 = __builtin_amdgcn_wmma_f32_16x16x32_bf16(false, a1.v, false, b0.v, (short)0, acc10, false, false);
    acc11 = __builtin_amdgcn_wmma_f32_16x16x32_bf16(false, a1.v, false, b1.v, (short)0, acc11, false, false);
  }
#pragma unroll
  for (int r = 0; r < 8; ++r) {
    const int row0 = tok0 + r + 8 * hi;
    Out[(size_t)row0        * Pp + col0] = f2bf(acc00[r]);
    Out[(size_t)row0        * Pp + col1] = f2bf(acc01[r]);
    Out[(size_t)(row0 + 16) * Pp + col0] = f2bf(acc10[r]);
    Out[(size_t)(row0 + 16) * Pp + col1] = f2bf(acc11[r]);
  }
}

// ---------------- fused GEMM + online logsumexp + target gather ----------------
// gridDim.y = nseg vocab segments; partial (max,sum,target,cluster) per segment.
// Reduction scratch overlays the A tile in LDS (A reads done before reuse).
__global__ __launch_bounds__(128) void gemm_lse(
    const unsigned short* __restrict__ A, int K,
    const unsigned short* __restrict__ Bw, const float* __restrict__ bias,
    int Vp, int nseg,
    const int* __restrict__ tgt, int base, int nvalid, int capture_cl,
    float* __restrict__ pm, float* __restrict__ ps, float* __restrict__ pt,
    float* __restrict__ pc0, float* __restrict__ pc1) {
  __shared__ __align__(16) char smem[32 * HDIM * 2];       // 64 KB, dual-use
  unsigned short* sA = (unsigned short*)smem;

  const int tid  = threadIdx.x;
  const int wave = tid >> 5, lane = tid & 31;
  const int hi = lane >> 4, lc = lane & 15, koff = hi * 8;
  const int tok0 = blockIdx.x * 32;

  {
    const u32x4* g = (const u32x4*)(A + (size_t)tok0 * K);
    u32x4* s = (u32x4*)sA;
    const int n = 4 * K;
    for (int i = tid; i < n; i += 128) s[i] = g[i];
  }
  __syncthreads();

  // stats index: idx = t*8+r  ->  output row = t*16 + r + 8*hi
  int tcap[16];
  float m[16], su[16], tl[16], c0v[16], c1v[16];
#pragma unroll
  for (int idx = 0; idx < 16; ++idx) {
    const int row = (idx >> 3) * 16 + (idx & 7) + 8 * hi;
    int t = tgt[tok0 + row] - base;
    tcap[idx] = (t >= 0 && t < nvalid) ? t : -1;
    m[idx] = NEG_BIG; su[idx] = 0.f; tl[idx] = NEG_BIG;
    c0v[idx] = NEG_BIG; c1v[idx] = NEG_BIG;
  }

#define ONLINE(idx, val, colv)                                                 \
  { float v_ = (val);                                                          \
    if (v_ > m[idx]) { su[idx] = su[idx] * __expf(m[idx] - v_) + 1.f; m[idx] = v_; } \
    else             { su[idx] += __expf(v_ - m[idx]); }                       \
    if ((colv) == tcap[idx]) tl[idx] = v_;                                     \
    if (capture_cl) {                                                          \
      if ((colv) == 20000) c0v[idx] = v_;                                      \
      else if ((colv) == 20001) c1v[idx] = v_; } }

  const int chunks = Vp >> 7;                   // 128-col chunks
  const int nlo = ((chunks * (int)blockIdx.y)       / nseg) << 7;
  const int nhi = ((chunks * ((int)blockIdx.y + 1)) / nseg) << 7;

  for (int n0 = nlo + wave * 32; n0 < nhi; n0 += 128) {
    const int col0 = n0 + lc, col1 = col0 + 16;
    const unsigned short* brow0 = Bw + (size_t)col0 * K;
    const unsigned short* brow1 = Bw + (size_t)col1 * K;
    if (col1 + 128 < Vp) {                      // global_prefetch_b8
      __builtin_prefetch(Bw + (size_t)(col0 + 128) * K, 0, 1);
      __builtin_prefetch(Bw + (size_t)(col1 + 128) * K, 0, 1);
    }
    v8f z = {0.f,0.f,0.f,0.f,0.f,0.f,0.f,0.f};
    v8f acc00 = z, acc01 = z, acc10 = z, acc11 = z;
#pragma unroll 2
    for (int k0 = 0; k0 < K; k0 += 32) {
      Frag a0, a1, b0, b1;
      a0.q[0] = *(const u32x4*)(sA + lc * K + k0 + koff);
      a0.q[1] = *(const u32x4*)(sA + lc * K + k0 + 16 + koff);
      a1.q[0] = *(const u32x4*)(sA + (lc + 16) * K + k0 + koff);
      a1.q[1] = *(const u32x4*)(sA + (lc + 16) * K + k0 + 16 + koff);
      b0.q[0] = *(const u32x4*)(brow0 + k0 + koff);
      b0.q[1] = *(const u32x4*)(brow0 + k0 + 16 + koff);
      b1.q[0] = *(const u32x4*)(brow1 + k0 + koff);
      b1.q[1] = *(const u32x4*)(brow1 + k0 + 16 + koff);
      acc00 = __builtin_amdgcn_wmma_f32_16x16x32_bf16(false, a0.v, false, b0.v, (short)0, acc00, false, false);
      acc01 = __builtin_amdgcn_wmma_f32_16x16x32_bf16(false, a0.v, false, b1.v, (short)0, acc01, false, false);
      acc10 = __builtin_amdgcn_wmma_f32_16x16x32_bf16(false, a1.v, false, b0.v, (short)0, acc10, false, false);
      acc11 = __builtin_amdgcn_wmma_f32_16x16x32_bf16(false, a1.v, false, b1.v, (short)0, acc11, false, false);
    }
    const float bb0 = bias[col0], bb1 = bias[col1];
#pragma unroll
    for (int r = 0; r < 8; ++r) {
      ONLINE(r,     acc00[r] + bb0, col0);
      ONLINE(r,     acc01[r] + bb1, col1);
      ONLINE(8 + r, acc10[r] + bb0, col0);
      ONLINE(8 + r, acc11[r] + bb1, col1);
    }
  }
#undef ONLINE

  // ---- reduce across lanes/waves; overlay stats scratch on the A tile ----
  __syncthreads();                               // all A reads complete
  float* rm  = (float*)smem;                     // [4][32][16] each
  float* rs  = rm  + 2048;
  float* rt  = rs  + 2048;
  float* rca = rt  + 2048;
  float* rcb = rca + 2048;
#pragma unroll
  for (int idx = 0; idx < 16; ++idx) {
    const int row = (idx >> 3) * 16 + (idx & 7) + 8 * hi;
    const int o = (wave * 32 + row) * 16 + lc;
    rm[o] = m[idx]; rs[o] = su[idx]; rt[o] = tl[idx];
    rca[o] = c0v[idx]; rcb[o] = c1v[idx];
  }
  __syncthreads();

  if (tid < 32) {
    const int row = tid;
    float M = NEG_BIG, S = 0.f, T = NEG_BIG, C0 = NEG_BIG, C1 = NEG_BIG;
    for (int w = 0; w < 4; ++w)
      for (int l = 0; l < 16; ++l) {
        const int o = (w * 32 + row) * 16 + l;
        float mm = rm[o], sv = rs[o];
        if (mm > M) { S = S * __expf(M - mm) + sv; M = mm; }
        else        { S += sv * __expf(mm - M); }
        T  = fmaxf(T,  rt[o]);
        C0 = fmaxf(C0, rca[o]);
        C1 = fmaxf(C1, rcb[o]);
      }
    const size_t o = (size_t)blockIdx.y * NTOK + tok0 + row;
    pm[o] = M; ps[o] = S; pt[o] = T;
    if (capture_cl) { pc0[o] = C0; pc1[o] = C1; }
  }
}

// ---------------- merge vocab segments -> per-token stats ----------------
__global__ void k_merge(const float* __restrict__ pm, const float* __restrict__ ps,
                        const float* __restrict__ pt,
                        const float* __restrict__ pc0, const float* __restrict__ pc1,
                        int nseg, int capture,
                        float* __restrict__ lse, float* __restrict__ tlog,
                        float* __restrict__ cl0, float* __restrict__ cl1) {
  int t = blockIdx.x * 256 + threadIdx.x;
  if (t >= NTOK) return;
  float M = NEG_BIG, S = 0.f, T = NEG_BIG, C0 = NEG_BIG, C1 = NEG_BIG;
  for (int s = 0; s < nseg; ++s) {
    size_t o = (size_t)s * NTOK + t;
    float mm = pm[o], sv = ps[o];
    if (mm > M) { S = S * __expf(M - mm) + sv; M = mm; }
    else        { S += sv * __expf(mm - M); }
    T = fmaxf(T, pt[o]);
    if (capture) { C0 = fmaxf(C0, pc0[o]); C1 = fmaxf(C1, pc1[o]); }
  }
  lse[t] = M + __logf(S);
  tlog[t] = T;
  if (capture) { cl0[t] = C0; cl1[t] = C1; }
}

// ---------------- final per-token NLL combine ----------------
__global__ void k_final(const int* __restrict__ tgt,
                        const float* __restrict__ lse0, const float* __restrict__ tl0,
                        const float* __restrict__ c0,  const float* __restrict__ c1,
                        const float* __restrict__ lse1, const float* __restrict__ tl1,
                        const float* __restrict__ lse2, const float* __restrict__ tl2,
                        float* __restrict__ out) {
  int t = blockIdx.x * 256 + threadIdx.x;
  if (t >= NTOK) return;
  int tg = tgt[t];
  float y = 0.f;
  if (tg >= 1 && tg < 20000)          y = lse0[t] - tl0[t];
  else if (tg >= 20000 && tg < 40000) y = (lse0[t] - c0[t]) + (lse1[t] - tl1[t]);
  else if (tg >= 40000)               y = (lse0[t] - c1[t]) + (lse2[t] - tl2[t]);
  out[t] = y;
}

// ---------------------------------------------------------------------------
extern "C" void kernel_launch(void* const* d_in, const int* in_sizes, int n_in,
                              void* d_out, int out_size, void* d_ws, size_t ws_size,
                              hipStream_t stream) {
  const float* x   = (const float*)d_in[0];
  const int*   tgt = (const int*)  d_in[1];
  const float* tw0 = (const float*)d_in[2];
  const float* tb0 = (const float*)d_in[3];
  const float* cw  = (const float*)d_in[4];
  const float* cb  = (const float*)d_in[5];
  const float* tw1 = (const float*)d_in[6];
  const float* tb1 = (const float*)d_in[7];
  const float* aw1 = (const float*)d_in[8];
  const float* tw2 = (const float*)d_in[9];
  const float* tb2 = (const float*)d_in[10];
  const float* aw2 = (const float*)d_in[11];

  const int T = NTOK, H = HDIM;
  const int V0p = 20096, V1p = 20096, V2p = 10368;  // padded to multiples of 128
  const int K1 = 384, K2 = 128;                     // p1=341, p2=113 zero-padded
  const int NSEG = 8;

  size_t off = 0;
  auto alloc = [&](size_t bytes) -> void* {
    off = (off + 255) & ~(size_t)255;
    void* p = (char*)d_ws + off;
    off += bytes;
    return p;
  };
  unsigned short* xb  = (unsigned short*)alloc((size_t)T   * H  * 2);
  unsigned short* w0b = (unsigned short*)alloc((size_t)V0p * H  * 2);
  float*          b0p = (float*)         alloc((size_t)V0p * 4);
  unsigned short* w1b = (unsigned short*)alloc((size_t)V1p * K1 * 2);
  float*          b1p = (float*)         alloc((size_t)V1p * 4);
  unsigned short* a1b = (unsigned short*)alloc((size_t)K1  * H  * 2);
  unsigned short* xt1 = (unsigned short*)alloc((size_t)T   * K1 * 2);
  unsigned short* w2b = (unsigned short*)alloc((size_t)V2p * K2 * 2);
  float*          b2p = (float*)         alloc((size_t)V2p * 4);
  unsigned short* a2b = (unsigned short*)alloc((size_t)K2  * H  * 2);
  unsigned short* xt2 = (unsigned short*)alloc((size_t)T   * K2 * 2);
  float* pm0 = (float*)alloc((size_t)NSEG * T * 4);
  float* ps0 = (float*)alloc((size_t)NSEG * T * 4);
  float* pt0 = (float*)alloc((size_t)NSEG * T * 4);
  float* pca = (float*)alloc((size_t)NSEG * T * 4);
  float* pcb = (float*)alloc((size_t)NSEG * T * 4);
  float* pm1 = (float*)alloc((size_t)NSEG * T * 4);
  float* ps1 = (float*)alloc((size_t)NSEG * T * 4);
  float* pt1 = (float*)alloc((size_t)NSEG * T * 4);
  float* pm2 = (float*)alloc((size_t)NSEG * T * 4);
  float* ps2 = (float*)alloc((size_t)NSEG * T * 4);
  float* pt2 = (float*)alloc((size_t)NSEG * T * 4);
  float* lse0 = (float*)alloc((size_t)T * 4);
  float* tl0  = (float*)alloc((size_t)T * 4);
  float* cc0  = (float*)alloc((size_t)T * 4);
  float* cc1  = (float*)alloc((size_t)T * 4);
  float* lse1 = (float*)alloc((size_t)T * 4);
  float* tl1  = (float*)alloc((size_t)T * 4);
  float* lse2 = (float*)alloc((size_t)T * 4);
  float* tl2  = (float*)alloc((size_t)T * 4);

  auto nb = [](long long n) { return (unsigned)((n + 255) / 256); };

  // Zero the padded bf16 weight buffers (pads must be 0 so padded dots = 0).
  k_zero16<<<nb((long long)V0p * H),  256, 0, stream>>>(w0b, (long long)V0p * H);
  k_zero16<<<nb((long long)V1p * K1), 256, 0, stream>>>(w1b, (long long)V1p * K1);
  k_zero16<<<nb((long long)V2p * K2), 256, 0, stream>>>(w2b, (long long)V2p * K2);
  k_zero16<<<nb((long long)K1  * H),  256, 0, stream>>>(a1b, (long long)K1  * H);
  k_zero16<<<nb((long long)K2  * H),  256, 0, stream>>>(a2b, (long long)K2  * H);

  // f32 -> bf16 conversions (bf16 weights ~60 MB stay resident in 192 MB L2).
  k_cvt<<<nb((long long)T * H),       256, 0, stream>>>(x,   xb,  T,     H,   H,  0);
  k_cvt<<<nb((long long)20000 * H),   256, 0, stream>>>(tw0, w0b, 20000, H,   H,  0);
  k_cvt<<<nb((long long)2 * H),       256, 0, stream>>>(cw,  w0b, 2,     H,   H,  20000);
  k_cvt<<<nb((long long)20000 * 341), 256, 0, stream>>>(tw1, w1b, 20000, 341, K1, 0);
  k_cvt<<<nb((long long)341 * H),     256, 0, stream>>>(aw1, a1b, 341,   H,   H,  0);
  k_cvt<<<nb((long long)10257 * 113), 256, 0, stream>>>(tw2, w2b, 10257, 113, K2, 0);
  k_cvt<<<nb((long long)113 * H),     256, 0, stream>>>(aw2, a2b, 113,   H,   H,  0);

  // Biases: pad with -1e30 so padded logits vanish in the softmax.
  k_fill_f32<<<nb(V0p),   256, 0, stream>>>(b0p, NEG_BIG, V0p);
  k_copy_f32<<<nb(20000), 256, 0, stream>>>(b0p, tb0, 20000);
  k_copy_f32<<<1,         256, 0, stream>>>(b0p + 20000, cb, 2);
  k_fill_f32<<<nb(V1p),   256, 0, stream>>>(b1p, NEG_BIG, V1p);
  k_copy_f32<<<nb(20000), 256, 0, stream>>>(b1p, tb1, 20000);
  k_fill_f32<<<nb(V2p),   256, 0, stream>>>(b2p, NEG_BIG, V2p);
  k_copy_f32<<<nb(10257), 256, 0, stream>>>(b2p, tb2, 10257);

  // Adapter projections (WMMA, bf16 out): xt1 = x @ aw1^T, xt2 = x @ aw2^T.
  gemm_xt<<<dim3(T / 32, K1 / 128), 128, 0, stream>>>(xb, H, a1b, K1, xt1);
  gemm_xt<<<dim3(T / 32, K2 / 128), 128, 0, stream>>>(xb, H, a2b, K2, xt2);

  // Head: fused GEMM + logsumexp, capture target + two cluster logits.
  gemm_lse<<<dim3(T / 32, NSEG), 128, 0, stream>>>(
      xb, H, w0b, b0p, V0p, NSEG, tgt, 0, 20000, 1, pm0, ps0, pt0, pca, pcb);
  k_merge<<<nb(T), 256, 0, stream>>>(pm0, ps0, pt0, pca, pcb, NSEG, 1,
                                     lse0, tl0, cc0, cc1);
  // Tail 1 (vocab 20000..40000, p1=341->K=384).
  gemm_lse<<<dim3(T / 32, NSEG), 128, 0, stream>>>(
      xt1, K1, w1b, b1p, V1p, NSEG, tgt, 20000, 20000, 0, pm1, ps1, pt1, nullptr, nullptr);
  k_merge<<<nb(T), 256, 0, stream>>>(pm1, ps1, pt1, nullptr, nullptr, NSEG, 0,
                                     lse1, tl1, nullptr, nullptr);
  // Tail 2 (vocab 40000..50257, p2=113->K=128).
  gemm_lse<<<dim3(T / 32, NSEG), 128, 0, stream>>>(
      xt2, K2, w2b, b2p, V2p, NSEG, tgt, 40000, 10257, 0, pm2, ps2, pt2, nullptr, nullptr);
  k_merge<<<nb(T), 256, 0, stream>>>(pm2, ps2, pt2, nullptr, nullptr, NSEG, 0,
                                     lse2, tl2, nullptr, nullptr);

  // Final combine -> per-token NLL [B,S].
  k_final<<<nb(T), 256, 0, stream>>>(tgt, lse0, tl0, cc0, cc1,
                                     lse1, tl1, lse2, tl2, (float*)d_out);
}
